// Nflow_73194832658780
// MI455X (gfx1250) — matrix operand order repaired and checked by reading, not compiled, not run on hardware
//
#include <hip/hip_runtime.h>
#include <hip/hip_bf16.h>
#include <math.h>

typedef __attribute__((ext_vector_type(16))) _Float16 v16h;
typedef __attribute__((ext_vector_type(8)))  _Float16 v8h;
typedef __attribute__((ext_vector_type(8)))  float    v8f;

union V16 { v16h v; v8h h[2]; };

#define HSTR 136      // LDS row stride in halves (272B = 68 dwords -> bank-skewed)
#define MT   128      // rows per block
#define BOUNDF 5.0f
#define SLOPEF 0.0001f

#if __has_builtin(__builtin_amdgcn_tensor_load_to_lds)
#define HAS_TDM 1
#else
#define HAS_TDM 0
#endif

// ---------------------------------------------------------------------------
// Prep: convert fp32 weights to transposed f16 in workspace.
// wsT layout (halves): wT1 @0, wT2 @16384, wT3 @32768, wT4(padded 16x128) @49152
// ---------------------------------------------------------------------------
__global__ void nflow_prep(const float* __restrict__ W1, const float* __restrict__ W2,
                           const float* __restrict__ W3, const float* __restrict__ W4,
                           _Float16* __restrict__ wsT) {
  int idx = blockIdx.x * blockDim.x + threadIdx.x;
  if (idx < 16384) {
    int n = idx >> 7, k = idx & 127;
    wsT[idx] = (_Float16)W1[k * 128 + n];
  } else if (idx < 32768) {
    int j = idx - 16384; int n = j >> 7, k = j & 127;
    wsT[idx] = (_Float16)W2[k * 128 + n];
  } else if (idx < 49152) {
    int j = idx - 32768; int n = j >> 7, k = j & 127;
    wsT[idx] = (_Float16)W3[k * 128 + n];
  } else if (idx < 51200) {
    int j = idx - 49152; int n = j >> 7, k = j & 127;
    wsT[idx] = (_Float16)((n < 14) ? W4[k * 14 + n] : 0.0f);
  }
}

// ---------------------------------------------------------------------------
// Weight staging: TDM (async tensor DMA) with padded LDS rows, or sync fallback
// ---------------------------------------------------------------------------
#if HAS_TDM
__device__ __forceinline__ void tdm_load_w(const _Float16* gsrc, _Float16* ldst,
                                           unsigned nrows) {
  typedef unsigned int u32x4 __attribute__((ext_vector_type(4)));
  typedef int          i32x4 __attribute__((ext_vector_type(4)));
  typedef int          i32x8 __attribute__((ext_vector_type(8)));
  unsigned long long ga = (unsigned long long)(uintptr_t)gsrc;
  unsigned int       la = (unsigned int)(uintptr_t)ldst;   // LDS byte offset

  // D# group 0: count=1 | lds_addr | global_addr[56:0] | type=2
  u32x4 g0;
  g0.x = 1u;
  g0.y = la;
  g0.z = (unsigned int)ga;
  g0.w = (unsigned int)(ga >> 32) | (2u << 30);

  // D# group 1: data_size=2B, pad 4 dwords every 64 dwords (128h row -> 136h),
  // tensor 128 x nrows (stride 128), tile = whole tensor.
  i32x8 g1;
  g1[0] = (int)((1u << 16) | (1u << 20) | (5u << 22) | (3u << 25));
  g1[1] = (int)(128u << 16);            // tensor_dim0[15:0]
  g1[2] = (int)(nrows << 16);           // tensor_dim1[15:0]
  g1[3] = (int)(128u << 16);            // tile_dim0
  g1[4] = (int)nrows;                   // tile_dim1
  g1[5] = 128;                          // tensor_dim0_stride (low 32)
  g1[6] = 0;
  g1[7] = 0;

  i32x4 gz = {0, 0, 0, 0};
#if defined(__clang_major__) && (__clang_major__ >= 23)
  i32x8 gz8 = {0, 0, 0, 0, 0, 0, 0, 0};
  __builtin_amdgcn_tensor_load_to_lds(g0, g1, gz, gz, gz8, 0);
#else
  __builtin_amdgcn_tensor_load_to_lds(g0, g1, gz, gz, 0);
#endif
}
#endif

// sync fallback: copy packed wT row-major [nrows][128] -> LDS [n*HSTR + k]
__device__ __forceinline__ void stage_w_sync(_Float16* dst, const _Float16* src,
                                             int t, int nrows) {
  const unsigned int* s = (const unsigned int*)src;       // 64 dwords per row
  int total = nrows * 64;
  for (int i = t; i < total; i += 256) {
    int n = i >> 6;
    int kk = (i & 63) << 1;
    *(unsigned int*)&dst[n * HSTR + kk] = s[i];
  }
}

// ---------------------------------------------------------------------------
// Fused MLP (WMMA) + rational-quadratic spline inverse
// ---------------------------------------------------------------------------
__global__ __launch_bounds__(256) void nflow_fused(
    const float* __restrict__ x, const float* __restrict__ z,
    const float* __restrict__ W0, const float* __restrict__ b0,
    const float* __restrict__ b1, const float* __restrict__ b2,
    const float* __restrict__ b3, const float* __restrict__ b4,
    const _Float16* __restrict__ wsT, float* __restrict__ out) {

  __shared__ _Float16 sA[2][MT * HSTR];   // activation ping-pong (f16)
  __shared__ _Float16 sW[2][128 * HSTR];  // weight tile double buffer (f16, [n][k])
  __shared__ float    sPhi[MT * 17];      // final head outputs (padded stride)

  const int t    = threadIdx.x;
  const int wave = t >> 5;                // 8 waves, each owns a 16-row strip
  const int lane = t & 31;
  const int col  = lane & 15;
  const int g    = lane >> 4;             // lane group (ISA A/B fragment split)
  const int rowbase = blockIdx.x * MT;

  // Kick off layer-1 weight DMA immediately; overlaps with layer-0 VALU work.
#if HAS_TDM
  if (wave == 0) tdm_load_w(wsT, &sW[0][0], 128u);
#else
  stage_w_sync(&sW[0][0], wsT, t, 128);
#endif

  // ---- Layer 0: h0 = relu(x * W0 + b0)  (1 -> 128, outer product) ----
  {
    int m  = t >> 1;
    int n0 = (t & 1) * 64;
    float xv = x[rowbase + m];
    #pragma unroll 8
    for (int j = 0; j < 64; ++j) {
      int n = n0 + j;
      float v = fmaf(xv, W0[n], b0[n]);
      sA[0][m * HSTR + n] = (_Float16)fmaxf(v, 0.0f);
    }
  }

#if HAS_TDM
  __builtin_amdgcn_s_wait_tensorcnt(0);
#endif
  __syncthreads();

  const float* biases[3] = { b1, b2, b3 };

  // ---- Layers 1..3: [128x128] GEMM via v_wmma_f32_16x16x32_f16 ----
  #pragma unroll 1
  for (int layer = 0; layer < 3; ++layer) {
    const _Float16* hin  = sA[layer & 1];
    _Float16*       hout = sA[(layer + 1) & 1];
    const _Float16* wcur = sW[layer & 1];

    // Async-stage the NEXT layer's weights into the other buffer while we
    // compute this layer (dest buffer was last read in layer-1; barrier done).
    {
      const _Float16* nw = wsT + (layer + 1) * 16384;
      unsigned nrows = (layer == 2) ? 16u : 128u;
#if HAS_TDM
      if (wave == 0) tdm_load_w(nw, &sW[(layer + 1) & 1][0], nrows);
#else
      stage_w_sync(&sW[(layer + 1) & 1][0], nw, t, (int)nrows);
#endif
    }

    // A fragments for this wave's 16-row strip, all 4 K-steps (register resident)
    V16 a[4];
    const int m = wave * 16 + col;
    #pragma unroll
    for (int kt = 0; kt < 4; ++kt) {
      const _Float16* p = &hin[m * HSTR + kt * 32 + 8 * g];
      a[kt].h[0] = *(const v8h*)(p);
      a[kt].h[1] = *(const v8h*)(p + 16);
    }

    const float* bptr = biases[layer];
    #pragma unroll
    for (int nt = 0; nt < 8; ++nt) {
      float bv = bptr[nt * 16 + col];     // C col depends only on lane
      v8f c = { bv, bv, bv, bv, bv, bv, bv, bv };
      #pragma unroll
      for (int kt = 0; kt < 4; ++kt) {
        V16 bf;
        const _Float16* q = &wcur[(nt * 16 + col) * HSTR + kt * 32 + 8 * g];
        bf.h[0] = *(const v8h*)(q);
        bf.h[1] = *(const v8h*)(q + 16);
        c = __builtin_amdgcn_wmma_f32_16x16x32_f16(
              false, a[kt].v, false, bf.v, (short)0, c, false, false);
      }
      const int rbase = wave * 16 + 8 * g;
      #pragma unroll
      for (int r = 0; r < 8; ++r)
        hout[(rbase + r) * HSTR + nt * 16 + col] = (_Float16)fmaxf(c[r], 0.0f);
    }

#if HAS_TDM
    __builtin_amdgcn_s_wait_tensorcnt(0);
#endif
    __syncthreads();
  }

  // ---- Layer 4: 128 -> 14 (padded to 16), one WMMA N-tile per wave ----
  {
    const _Float16* hin  = sA[1];   // layer-3 output lives in sA[1]
    const _Float16* wcur = sW[1];   // wT4 staged into buffer 1 during layer 2
    V16 a[4];
    const int m = wave * 16 + col;
    #pragma unroll
    for (int kt = 0; kt < 4; ++kt) {
      const _Float16* p = &hin[m * HSTR + kt * 32 + 8 * g];
      a[kt].h[0] = *(const v8h*)(p);
      a[kt].h[1] = *(const v8h*)(p + 16);
    }
    float bv = (col < 14) ? b4[col] : 0.0f;
    v8f c = { bv, bv, bv, bv, bv, bv, bv, bv };
    #pragma unroll
    for (int kt = 0; kt < 4; ++kt) {
      V16 bf;
      const _Float16* q = &wcur[col * HSTR + kt * 32 + 8 * g];
      bf.h[0] = *(const v8h*)(q);
      bf.h[1] = *(const v8h*)(q + 16);
      c = __builtin_amdgcn_wmma_f32_16x16x32_f16(
            false, a[kt].v, false, bf.v, (short)0, c, false, false);
    }
    const int rbase = wave * 16 + 8 * g;
    #pragma unroll
    for (int r = 0; r < 8; ++r)
      sPhi[(rbase + r) * 17 + col] = c[r];
  }
  __syncthreads();

  // ---- Rational-quadratic spline inverse (Durkan et al.), one row/thread ----
  if (t < MT) {
    float phi[14];
    #pragma unroll
    for (int i = 0; i < 14; ++i) phi[i] = sPhi[t * 17 + i];
    float zv = z[rowbase + t];

    float wmx = phi[0];
    #pragma unroll
    for (int i = 1; i < 5; ++i) wmx = fmaxf(wmx, phi[i]);
    float we[5], wsum = 0.0f;
    #pragma unroll
    for (int i = 0; i < 5; ++i) { we[i] = __expf(phi[i] - wmx); wsum += we[i]; }
    float wscale = (2.0f * BOUNDF) / wsum;

    float hmx = phi[5];
    #pragma unroll
    for (int i = 1; i < 5; ++i) hmx = fmaxf(hmx, phi[5 + i]);
    float he[5], hsum = 0.0f;
    #pragma unroll
    for (int i = 0; i < 5; ++i) { he[i] = __expf(phi[5 + i] - hmx); hsum += he[i]; }
    float hscale = (2.0f * BOUNDF) / hsum;

    float xk[6], yk[6];
    xk[0] = -BOUNDF; yk[0] = -BOUNDF;
    #pragma unroll
    for (int i = 0; i < 5; ++i) {
      xk[i + 1] = xk[i] + we[i] * wscale;
      yk[i + 1] = yk[i] + he[i] * hscale;
    }
    float dk[6];
    dk[0] = 1.0f; dk[5] = 1.0f;
    #pragma unroll
    for (int i = 0; i < 4; ++i) {
      float v = phi[10 + i];
      float sp = (v > 20.0f) ? v : log1pf(__expf(v));   // softplus
      dk[i + 1] = SLOPEF + sp;
    }

    float zc = fminf(fmaxf(zv, -BOUNDF), BOUNDF);
    int k = 0;
    #pragma unroll
    for (int i = 1; i <= 4; ++i) k += (zc >= yk[i]) ? 1 : 0;

    float x0 = xk[k], x1 = xk[k + 1];
    float y0 = yk[k], y1 = yk[k + 1];
    float d0 = dk[k], d1 = dk[k + 1];
    float dx = x1 - x0, dy = y1 - y0;
    float s  = dy / dx;
    float tt = zc - y0;
    float mm = d0 + d1 - 2.0f * s;
    float aa = dy * (s - d0) + tt * mm;
    float bb = dy * d0 - tt * mm;
    float cc = -s * tt;
    float disc = fmaxf(bb * bb - 4.0f * aa * cc, 0.0f);
    float xi = 2.0f * cc / (-bb - sqrtf(disc));
    float xin = x0 + xi * dx;

    out[rowbase + t] = (fabsf(zv) >= BOUNDF) ? zv : xin;
  }
}

// ---------------------------------------------------------------------------
extern "C" void kernel_launch(void* const* d_in, const int* in_sizes, int n_in,
                              void* d_out, int out_size, void* d_ws, size_t ws_size,
                              hipStream_t stream) {
  const float* x  = (const float*)d_in[0];
  const float* z  = (const float*)d_in[1];
  const float* W0 = (const float*)d_in[2];
  const float* b0 = (const float*)d_in[3];
  const float* W1 = (const float*)d_in[4];
  const float* b1 = (const float*)d_in[5];
  const float* W2 = (const float*)d_in[6];
  const float* b2 = (const float*)d_in[7];
  const float* W3 = (const float*)d_in[8];
  const float* b3 = (const float*)d_in[9];
  const float* W4 = (const float*)d_in[10];
  const float* b4 = (const float*)d_in[11];
  _Float16* wsT = (_Float16*)d_ws;
  float* out = (float*)d_out;
  const int N = in_sizes[0];

  nflow_prep<<<(51200 + 255) / 256, 256, 0, stream>>>(W1, W2, W3, W4, wsT);
  nflow_fused<<<N / MT, 256, 0, stream>>>(x, z, W0, b0, b1, b2, b3, b4, wsT, out);
}
